// Decoder_8804682957186
// MI455X (gfx1250) — matrix operand order repaired and checked by reading, not compiled, not run on hardware
//
#include <hip/hip_runtime.h>

// ---------------------------------------------------------------------------
// MI455X (gfx1250) Tacotron2-style decoder.
// - Persistent 192-block kernel, LDS-resident bf16 LSTM weights, device-wide
//   barriers (3 per step x 800 steps).
// - v_wmma_f32_16x16x32_bf16 for prenet / LSTM gates / postnet conv-as-GEMM.
// - All fragment loads are branch-free vector loads (inputs pre-padded/packed).
// - fp32 for gates nonlinearity, attention, softmax, output projections.
// ---------------------------------------------------------------------------

typedef __attribute__((ext_vector_type(16))) __bf16 v16bf;
typedef __attribute__((ext_vector_type(8)))  __bf16 v8bf;
typedef __attribute__((ext_vector_type(8)))  float  v8f;

#define B_   64
#define T_   800
#define S_   200
#define M_   80
#define PRE_ 256
#define H_   1024
#define E_   512
#define A_   128
#define LF_  32
#define PF_  512
#define TP_  840     // time-padded rows for postnet activations (2 front, rest tail)

#define GS    68     // padded LDS stride for the 64x64 gate staging tile
#define NBLK  192    // persistent blocks: 64 LSTM1 + 64 LSTM2 + 64 attention
#define NW1   (4 * 56 * 512)   // LSTM1 per-block LDS weight image elems (114688)
#define NW2   (4 * 64 * 512)   // LSTM2 per-block LDS weight image elems (131072)

// ---------------------------------------------------------------------------
// device-wide barrier (agent scope, s_sleep spin)
// ---------------------------------------------------------------------------
__device__ __forceinline__ void gbar(int* cnt, int* gen, int nb) {
  __threadfence();
  __syncthreads();
  if (threadIdx.x == 0) {
    int g = __hip_atomic_load(gen, __ATOMIC_RELAXED, __HIP_MEMORY_SCOPE_AGENT);
    int prev = __hip_atomic_fetch_add(cnt, 1, __ATOMIC_ACQ_REL, __HIP_MEMORY_SCOPE_AGENT);
    if (prev == nb - 1) {
      __hip_atomic_store(cnt, 0, __ATOMIC_RELAXED, __HIP_MEMORY_SCOPE_AGENT);
      __hip_atomic_fetch_add(gen, 1, __ATOMIC_RELEASE, __HIP_MEMORY_SCOPE_AGENT);
    } else {
      while (__hip_atomic_load(gen, __ATOMIC_ACQUIRE, __HIP_MEMORY_SCOPE_AGENT) == g) {
        __builtin_amdgcn_s_sleep(2);
      }
    }
  }
  __syncthreads();
}

// ---------------------------------------------------------------------------
// prep kernels
// ---------------------------------------------------------------------------
// per-block LDS fragment image for LSTM1 ([blk 0..63][NW1])
__global__ void k_img_w1(const float* __restrict__ wih, const float* __restrict__ whh,
                         __bf16* __restrict__ img) {
  size_t gid = (size_t)blockIdx.x * 256 + threadIdx.x;
  if (gid >= (size_t)64 * NW1) return;
  const int blkj = (int)(gid / NW1);
  const int idx  = (int)(gid % NW1);
  const int e = idx & 15;
  const int lane = (idx >> 4) & 31;
  const int kb = (idx >> 9) % 56;
  const int ntile = idx / (56 * 512);
  const int row = ntile * H_ + (blkj << 4) + (lane & 15);
  const int k = (kb << 5) + ((lane >> 4) << 4) + e;
  float v = (k < 768) ? wih[(size_t)row * 768 + k] : whh[(size_t)row * 1024 + (k - 768)];
  img[gid] = (__bf16)v;
}

// per-block LDS fragment image for LSTM2 ([blk 0..63][NW2])
__global__ void k_img_w2(const float* __restrict__ wih, const float* __restrict__ whh,
                         __bf16* __restrict__ img) {
  size_t gid = (size_t)blockIdx.x * 256 + threadIdx.x;
  if (gid >= (size_t)64 * NW2) return;
  const int blkj = (int)(gid / NW2);
  const int idx  = (int)(gid % NW2);
  const int e = idx & 15;
  const int lane = (idx >> 4) & 31;
  const int kb = (idx >> 9) & 63;
  const int ntile = idx / (64 * 512);
  const int row = ntile * H_ + (blkj << 4) + (lane & 15);
  const int k = (kb << 5) + ((lane >> 4) << 4) + e;
  float v = (k < 1024) ? wih[(size_t)row * 1024 + k] : whh[(size_t)row * 1024 + (k - 1024)];
  img[gid] = (__bf16)v;
}

// shifted, zero-padded bf16 mel matrix [B*T][96]
__global__ void k_build_mws(const float* __restrict__ mels, __bf16* __restrict__ mws) {
  int idx = blockIdx.x * 256 + threadIdx.x;
  if (idx >= 51200 * 96) return;
  const int k = idx % 96, row = idx / 96;
  const int bb = row / T_, tt = row % T_;
  float v = (k < M_ && tt > 0) ? mels[((size_t)bb * T_ + (tt - 1)) * M_ + k] : 0.f;
  mws[idx] = (__bf16)v;
}

// pre_w0 padded to [256][96] bf16
__global__ void k_pad_w0(const float* __restrict__ w, __bf16* __restrict__ d) {
  int idx = blockIdx.x * 256 + threadIdx.x;
  if (idx >= 256 * 96) return;
  const int n = idx / 96, k = idx % 96;
  d[idx] = (__bf16)((k < M_) ? w[(size_t)n * M_ + k] : 0.f);
}

__global__ void k_cvt_bf16(const float* __restrict__ s, __bf16* __restrict__ d, int n) {
  int i = blockIdx.x * 256 + threadIdx.x;
  if (i < n) d[i] = (__bf16)s[i];
}

// postnet weight repack: dst[co][kk*Cp+ci], zero-padded in co and ci
__global__ void k_pack_pnw(const float* __restrict__ src, __bf16* __restrict__ dst,
                           int Ci, int Cp, int Co, int Cop) {
  const int K = 5 * Cp;
  size_t tot = (size_t)Cop * K;
  size_t idx = (size_t)blockIdx.x * 256 + threadIdx.x;
  if (idx >= tot) return;
  const int co = (int)(idx / K);
  const int r = (int)(idx % K);
  const int kk = r / Cp, ci = r % Cp;
  float v = (co < Co && ci < Ci) ? src[((size_t)co * Ci + ci) * 5 + kk] : 0.f;
  dst[idx] = (__bf16)v;
}

// mel_outs (fp32 [b][t][80]) -> time-padded bf16 [b][TP_][128] (buffer pre-zeroed)
__global__ void k_mel2bf_pad(const float* __restrict__ mel, __bf16* __restrict__ d) {
  int idx = blockIdx.x * 256 + threadIdx.x;
  if (idx >= B_ * T_ * M_) return;
  const int ci = idx % M_;
  const int bt = idx / M_;
  const int t = bt % T_, b = bt / T_;
  d[((size_t)b * TP_ + (t + 2)) * 128 + ci] = (__bf16)mel[idx];
}

__global__ void k_vcache(const float* __restrict__ enc, const float* __restrict__ wv,
                         float* __restrict__ vc) {
  int idx = blockIdx.x * 256 + threadIdx.x;
  if (idx >= B_ * S_ * A_) return;
  int a  = idx & (A_ - 1);
  int bs = idx >> 7;
  const float* ep = enc + (size_t)bs * E_;
  const float* wr = wv + (size_t)a * E_;
  float acc = 0.f;
  for (int e = 0; e < E_; ++e) acc += ep[e] * wr[e];
  vc[idx] = acc;
}

// ---------------------------------------------------------------------------
// prenet GEMM 1: relu(mels_ws @ pre_w0^T)  rows=B*T, K=96(padded), N=256
// ---------------------------------------------------------------------------
__global__ __launch_bounds__(256) void k_prenet1(const __bf16* __restrict__ MWS,
                                                 const __bf16* __restrict__ W0,
                                                 __bf16* __restrict__ X0) {
  const int mBase = blockIdx.x * 64;
  const int nBase = blockIdx.y * 64;
  const int tid = threadIdx.x, wave = tid >> 5, lane = tid & 31;
  const int lm = lane & 15, lh = lane >> 4;
#pragma unroll
  for (int rep = 0; rep < 2; ++rep) {
    const int wt = wave + (rep << 3);
    const int mi = wt & 3, ni = wt >> 2;
    const int m = mBase + (mi << 4) + lm;
    const int n = nBase + (ni << 4) + lm;
    v8f acc;
#pragma unroll
    for (int e = 0; e < 8; ++e) acc[e] = 0.f;
#pragma unroll
    for (int kb = 0; kb < 3; ++kb) {
      const int kh = (kb << 5) + (lh << 4);
      v8bf lo = *(const v8bf*)(MWS + (size_t)m * 96 + kh);
      v8bf hi = *(const v8bf*)(MWS + (size_t)m * 96 + kh + 8);
      v16bf afrag;
#pragma unroll
      for (int e = 0; e < 8; ++e) { afrag[e] = lo[e]; afrag[8 + e] = hi[e]; }
      const int k0 = (kb << 5) + (lh << 4);
      v16bf bfrag = *(const v16bf*)(W0 + (size_t)n * 96 + k0);
      acc = __builtin_amdgcn_wmma_f32_16x16x32_bf16(false, afrag, false, bfrag,
                                                    (short)0, acc, false, false);
    }
    const int mb = mBase + (mi << 4) + (lh << 3);
#pragma unroll
    for (int r = 0; r < 8; ++r)
      X0[(size_t)(mb + r) * PRE_ + n] = (__bf16)fmaxf(acc[r], 0.f);
  }
}

// ---------------------------------------------------------------------------
// prenet GEMM 2: relu(X0 @ pre_w1^T), output time-major xpre[t][b][256]
// ---------------------------------------------------------------------------
__global__ __launch_bounds__(256) void k_prenet2(const __bf16* __restrict__ X0,
                                                 const __bf16* __restrict__ W1,
                                                 __bf16* __restrict__ xpre) {
  const int mBase = blockIdx.x * 64;
  const int nBase = blockIdx.y * 64;
  const int tid = threadIdx.x, wave = tid >> 5, lane = tid & 31;
  const int lm = lane & 15, lh = lane >> 4;
#pragma unroll
  for (int rep = 0; rep < 2; ++rep) {
    const int wt = wave + (rep << 3);
    const int mi = wt & 3, ni = wt >> 2;
    const int m = mBase + (mi << 4) + lm;
    const int n = nBase + (ni << 4) + lm;
    v8f acc;
#pragma unroll
    for (int e = 0; e < 8; ++e) acc[e] = 0.f;
    for (int kb = 0; kb < 8; ++kb) {
      const int kh = (kb << 5) + (lh << 4);
      v8bf lo = *(const v8bf*)(X0 + (size_t)m * PRE_ + kh);
      v8bf hi = *(const v8bf*)(X0 + (size_t)m * PRE_ + kh + 8);
      v16bf afrag;
#pragma unroll
      for (int e = 0; e < 8; ++e) { afrag[e] = lo[e]; afrag[8 + e] = hi[e]; }
      const int k0 = (kb << 5) + (lh << 4);
      v16bf bfrag = *(const v16bf*)(W1 + (size_t)n * PRE_ + k0);
      acc = __builtin_amdgcn_wmma_f32_16x16x32_bf16(false, afrag, false, bfrag,
                                                    (short)0, acc, false, false);
    }
    const int mb = mBase + (mi << 4) + (lh << 3);
#pragma unroll
    for (int r = 0; r < 8; ++r) {
      const int row = mb + r;
      const int bbr = row / T_, ttr = row % T_;
      xpre[((size_t)ttr * B_ + bbr) * PRE_ + n] = (__bf16)fmaxf(acc[r], 0.f);
    }
  }
}

// ---------------------------------------------------------------------------
// LSTM gate phase: 64x(64 gate rows) GEMM from LDS-resident weights + cell.
// role 0: A = [x_t | ctx | h1_old], K=1792.  role 1: A = [h1_new | h2_old], K=2048.
// ---------------------------------------------------------------------------
__device__ __forceinline__ void lstm_phase(int role, int tid,
    const __bf16* __restrict__ s0, const __bf16* __restrict__ s1,
    const __bf16* __restrict__ s2,
    const __bf16* __restrict__ wlds, float* gates_s, float* c_s,
    const float* __restrict__ bA, const float* __restrict__ bB,
    __bf16* __restrict__ hout, int jBase, int KB) {
  const int wave = tid >> 5, lane = tid & 31;
  const int lm = lane & 15, lh = lane >> 4;
#pragma unroll
  for (int rep = 0; rep < 2; ++rep) {
    const int wt = wave + (rep << 3);
    const int mi = wt & 3, ni = wt >> 2;
    const int b = (mi << 4) + lm;
    v8f acc;
#pragma unroll
    for (int e = 0; e < 8; ++e) acc[e] = 0.f;
    for (int kb = 0; kb < KB; ++kb) {
      const int kh = (kb << 5) + (lh << 4);
      const __bf16* p0;
      const __bf16* p1;
      if (role == 0) {
        p0 = (kh < 256) ? (s0 + b * PRE_ + kh)
           : (kh < 768) ? (s1 + b * E_ + (kh - 256))
                        : (s2 + b * H_ + (kh - 768));
        const int k2 = kh + 8;
        p1 = (k2 < 256) ? (s0 + b * PRE_ + k2)
           : (k2 < 768) ? (s1 + b * E_ + (k2 - 256))
                        : (s2 + b * H_ + (k2 - 768));
      } else {
        p0 = (kh < H_) ? (s0 + b * H_ + kh) : (s1 + b * H_ + (kh - H_));
        const int k2 = kh + 8;
        p1 = (k2 < H_) ? (s0 + b * H_ + k2) : (s1 + b * H_ + (k2 - H_));
      }
      v8bf lo = *(const v8bf*)p0;
      v8bf hi = *(const v8bf*)p1;
      v16bf afrag;
#pragma unroll
      for (int e = 0; e < 8; ++e) { afrag[e] = lo[e]; afrag[8 + e] = hi[e]; }
      v16bf bfrag = *(const v16bf*)(wlds + (((size_t)(ni * KB + kb) * 32 + lane) << 4));
      acc = __builtin_amdgcn_wmma_f32_16x16x32_bf16(false, afrag, false, bfrag,
                                                    (short)0, acc, false, false);
    }
    const int n  = (ni << 4) + lm;
    const int mb = (mi << 4) + (lh << 3);
#pragma unroll
    for (int r = 0; r < 8; ++r) gates_s[(mb + r) * GS + n] = acc[r];
  }
  __syncthreads();
  for (int idx = tid; idx < B_ * 16; idx += 256) {
    const int b = idx >> 4, jl = idx & 15;
    const int j = jBase + jl;
    float gi = gates_s[b * GS + jl]      + bA[j]          + bB[j];
    float gf = gates_s[b * GS + 16 + jl] + bA[H_ + j]     + bB[H_ + j];
    float gg = gates_s[b * GS + 32 + jl] + bA[2 * H_ + j] + bB[2 * H_ + j];
    float go = gates_s[b * GS + 48 + jl] + bA[3 * H_ + j] + bB[3 * H_ + j];
    float si = 1.f / (1.f + __expf(-gi));
    float sf = 1.f / (1.f + __expf(-gf));
    float so = 1.f / (1.f + __expf(-go));
    float c = sf * c_s[idx] + si * tanhf(gg);
    c_s[idx] = c;
    hout[b * H_ + j] = (__bf16)(so * tanhf(c));
  }
  __syncthreads();
}

// ---------------------------------------------------------------------------
// persistent decode kernel: 192 blocks, 800 steps, 3 device barriers per step
// ---------------------------------------------------------------------------
__global__ __launch_bounds__(256) void k_decode(
    const __bf16* __restrict__ W1IMG, const __bf16* __restrict__ W2IMG,
    const __bf16* __restrict__ xpre,
    const float* __restrict__ bih1, const float* __restrict__ bhh1,
    const float* __restrict__ bih2, const float* __restrict__ bhh2,
    __bf16* h1b0, __bf16* h1b1, __bf16* h2b0, __bf16* h2b1,
    __bf16* ctxbf, float* ctxf,
    const float* __restrict__ enc, const unsigned char* __restrict__ emask,
    const float* __restrict__ vcache,
    const float* __restrict__ wq, const float* __restrict__ bq,
    const float* __restrict__ locw, const float* __restrict__ locproj,
    const float* __restrict__ vvec,
    const float* __restrict__ wf, const float* __restrict__ bfb,
    const float* __restrict__ wsv, const float* __restrict__ bsv,
    float* __restrict__ out_mel, float* __restrict__ out_stop,
    float* __restrict__ out_aw,
    int* bar_cnt, int* bar_gen) {
  extern __shared__ char smem[];
  const int blk = blockIdx.x, tid = threadIdx.x;
  const int role = (blk < 64) ? 0 : (blk < 128) ? 1 : 2;

  __bf16* wlds = nullptr;
  float* gates_s = nullptr;
  float* c_s = nullptr;
  float *cum_s = nullptr, *loc_s = nullptr, *q_s = nullptr;
  float *e_s = nullptr, *aw_s = nullptr, *red_s = nullptr;
  int jBase = 0, batt = 0;

  if (role < 2) {
    const int nw = (role == 0) ? NW1 : NW2;
    jBase = (blk & 63) << 4;
    wlds = (__bf16*)smem;
    gates_s = (float*)(smem + (size_t)nw * 2);
    c_s = gates_s + B_ * GS;
    // straight vector copy of the prebuilt per-block fragment image into LDS
    const v16bf* src = (const v16bf*)(((role == 0) ? W1IMG : W2IMG) + (size_t)(blk & 63) * nw);
    v16bf* dst = (v16bf*)wlds;
    for (int i = tid; i < (nw >> 4); i += 256) dst[i] = src[i];
    for (int i = tid; i < B_ * 16; i += 256) c_s[i] = 0.f;
  } else {
    batt = blk - 128;
    cum_s = (float*)smem;            // 200 (+pad)
    loc_s = cum_s + 256;             // 32*200
    q_s   = loc_s + LF_ * S_;        // 128
    e_s   = q_s + A_;                // 200
    aw_s  = e_s + S_;                // 200
    red_s = aw_s + S_;               // 1
    for (int i = tid; i < S_; i += 256) cum_s[i] = 0.f;
  }
  __syncthreads();

  for (int t = 0; t < T_; ++t) {
    const __bf16* h1r = (t & 1) ? h1b1 : h1b0;
    __bf16*       h1w = (t & 1) ? h1b0 : h1b1;
    const __bf16* h2r = (t & 1) ? h2b1 : h2b0;
    __bf16*       h2w = (t & 1) ? h2b0 : h2b1;

    if (role == 0) {
      if (t + 1 < T_)   // warm L2/L0 for next step's prenet activations
        __builtin_prefetch(xpre + (size_t)(t + 1) * B_ * PRE_ + tid * 64, 0, 1);
      lstm_phase(0, tid, xpre + (size_t)t * B_ * PRE_, ctxbf, h1r,
                 wlds, gates_s, c_s, bih1, bhh1, h1w, jBase, 56);
    }
    gbar(bar_cnt, bar_gen, NBLK);

    if (role == 1)
      lstm_phase(1, tid, h1w, h2r, nullptr,
                 wlds, gates_s, c_s, bih2, bhh2, h2w, jBase, 64);
    gbar(bar_cnt, bar_gen, NBLK);

    if (role == 2) {
      const int b = batt;
      const __bf16* h2 = h2w;
      if (tid < A_) {                                  // q = h2 @ wq^T + bq
        float acc = bq[tid];
        const float* wr = wq + (size_t)tid * H_;
        for (int h = 0; h < H_; ++h) acc += (float)h2[b * H_ + h] * wr[h];
        q_s[tid] = acc;
      }
      __syncthreads();
      for (int idx = tid; idx < LF_ * S_; idx += 256) {  // location conv (pre-update cum)
        const int f = idx / S_, s = idx - f * S_;
        float acc = 0.f;
        const float* lw = locw + f * 31;
#pragma unroll
        for (int k = 0; k < 31; ++k) {
          const int sp = s + k - 15;
          if (sp >= 0 && sp < S_) acc += lw[k] * cum_s[sp];
        }
        loc_s[idx] = acc;
      }
      __syncthreads();
      for (int s = tid; s < S_; s += 256) {            // energies
        const float* vc = vcache + ((size_t)(b * S_ + s)) * A_;
        float e = 0.f;
        for (int a = 0; a < A_; ++a) {
          float l = 0.f;
          const float* lp = locproj + a * LF_;
#pragma unroll
          for (int f = 0; f < LF_; ++f) l += loc_s[f * S_ + s] * lp[f];
          e += tanhf(q_s[a] + vc[a] + l) * vvec[a];
        }
        if (emask[b * S_ + s]) e = -1.0e9f;
        e_s[s] = e;
      }
      __syncthreads();
      if (tid == 0) {                                  // softmax normalizer
        float mx = -3.4e38f;
        for (int s = 0; s < S_; ++s) mx = fmaxf(mx, e_s[s]);
        float sum = 0.f;
        for (int s = 0; s < S_; ++s) { float w = __expf(e_s[s] - mx); e_s[s] = w; sum += w; }
        red_s[0] = 1.f / sum;
      }
      __syncthreads();
      const float rinv = red_s[0];
      for (int s = tid; s < S_; s += 256) {
        const float aw = e_s[s] * rinv;
        aw_s[s] = aw;
        cum_s[s] += aw;
        out_aw[((size_t)b * T_ + t) * S_ + s] = aw;
      }
      __syncthreads();
      for (int e0 = tid; e0 < E_; e0 += 256) {         // ctx = aw @ enc
        float acc = 0.f;
        const float* ep = enc + ((size_t)b * S_) * E_ + e0;
        for (int s = 0; s < S_; ++s) acc += aw_s[s] * ep[(size_t)s * E_];
        ctxf[b * E_ + e0] = acc;
        ctxbf[b * E_ + e0] = (__bf16)acc;
      }
      __syncthreads();
      if (tid < M_) {                                  // mel frame
        float acc = bfb[tid];
        const float* wr = wf + (size_t)tid * (H_ + E_);
        for (int h = 0; h < H_; ++h) acc += (float)h2[b * H_ + h] * wr[h];
        for (int e0 = 0; e0 < E_; ++e0) acc += ctxf[b * E_ + e0] * wr[H_ + e0];
        out_mel[((size_t)b * T_ + t) * M_ + tid] = acc;
      } else if (tid == 128) {                         // stop token
        float acc = bsv[0];
        for (int h = 0; h < H_; ++h) acc += (float)h2[b * H_ + h] * wsv[h];
        for (int e0 = 0; e0 < E_; ++e0) acc += ctxf[b * E_ + e0] * wsv[H_ + e0];
        out_stop[(size_t)b * T_ + t] = acc;
      }
    }
    gbar(bar_cnt, bar_gen, NBLK);
  }
}

// ---------------------------------------------------------------------------
// postnet conv-as-GEMM with K = kk*Cp + ci ordering, time-padded inputs.
// in: bf16 [b][TP_][Cp], W: bf16 [Cop][5*Cp]; branch-free vector loads.
// ---------------------------------------------------------------------------
__global__ __launch_bounds__(256) void k_post(
    const __bf16* __restrict__ in, const __bf16* __restrict__ W,
    const float* __restrict__ bias, const float* __restrict__ gamma,
    const float* __restrict__ beta,
    int Cp, int l2cp, int Co, int KB, int dotanh,
    __bf16* __restrict__ outbf, float* __restrict__ outf) {
  const int tBase = blockIdx.x * 64;
  const int nBase = blockIdx.y * 64;
  const int b = blockIdx.z;
  const float inv = rsqrtf(1.f + 1e-5f);
  const int Kdim = 5 * Cp;
  const int cmask = Cp - 1;
  const int tid = threadIdx.x, wave = tid >> 5, lane = tid & 31;
  const int lm = lane & 15, lh = lane >> 4;
#pragma unroll
  for (int rep = 0; rep < 2; ++rep) {
    const int wt = wave + (rep << 3);
    const int mi = wt & 3, ni = wt >> 2;
    const int trow = tBase + (mi << 4) + lm;     // padded-buffer base row
    const int n = nBase + (ni << 4) + lm;
    const __bf16* inb = in + (size_t)b * TP_ * Cp;
    v8f acc;
#pragma unroll
    for (int e = 0; e < 8; ++e) acc[e] = 0.f;
    for (int kb = 0; kb < KB; ++kb) {
      const int kh = (kb << 5) + (lh << 4);
      const int kk0 = kh >> l2cp, ci0 = kh & cmask;
      const int kh2 = kh + 8;
      const int kk1 = kh2 >> l2cp, ci1 = kh2 & cmask;
      v8bf lo = *(const v8bf*)(inb + (size_t)(trow + kk0) * Cp + ci0);
      v8bf hi = *(const v8bf*)(inb + (size_t)(trow + kk1) * Cp + ci1);
      v16bf afrag;
#pragma unroll
      for (int e = 0; e < 8; ++e) { afrag[e] = lo[e]; afrag[8 + e] = hi[e]; }
      const int k0 = (kb << 5) + (lh << 4);
      v16bf bfrag = *(const v16bf*)(W + (size_t)n * Kdim + k0);
      acc = __builtin_amdgcn_wmma_f32_16x16x32_bf16(false, afrag, false, bfrag,
                                                    (short)0, acc, false, false);
    }
    const int mb = tBase + (mi << 4) + (lh << 3);
    if (n < Co) {
#pragma unroll
      for (int r = 0; r < 8; ++r) {
        const int t = mb + r;
        if (t < T_) {
          float y = (acc[r] + bias[n]) * inv * gamma[n] + beta[n];
          if (dotanh) y = tanhf(y);
          if (outf) outf[((size_t)b * T_ + t) * M_ + n] = y;
          else      outbf[((size_t)b * TP_ + (t + 2)) * PF_ + n] = (__bf16)y;
        }
      }
    }
  }
}

// ---------------------------------------------------------------------------
// decoder-mask epilogue
// ---------------------------------------------------------------------------
__global__ void k_mask(const unsigned char* __restrict__ dm,
                       float* __restrict__ out_res, float* __restrict__ out_stop,
                       float* __restrict__ out_aw) {
  int idx = blockIdx.x * 256 + threadIdx.x;
  if (idx >= B_ * T_) return;
  if (dm[idx]) {
    out_stop[idx] = 1000.f;
    float* r = out_res + (size_t)idx * M_;
    for (int m = 0; m < M_; ++m) r[m] = 0.f;
    float* a = out_aw + (size_t)idx * S_;
    for (int s = 0; s < S_; ++s) a[s] = 0.f;
  }
}

// ---------------------------------------------------------------------------
// host launcher
// ---------------------------------------------------------------------------
extern "C" void kernel_launch(void* const* d_in, const int* in_sizes, int n_in,
                              void* d_out, int out_size, void* d_ws, size_t ws_size,
                              hipStream_t stream) {
  (void)in_sizes; (void)n_in; (void)out_size; (void)ws_size;
  const float* enc   = (const float*)d_in[0];
  const float* mels  = (const float*)d_in[1];
  const unsigned char* emask = (const unsigned char*)d_in[2];
  const unsigned char* dmask = (const unsigned char*)d_in[3];
  const float* pre_w0 = (const float*)d_in[4];
  const float* pre_w1 = (const float*)d_in[5];
  const float* wih1 = (const float*)d_in[6];
  const float* whh1 = (const float*)d_in[7];
  const float* bih1 = (const float*)d_in[8];
  const float* bhh1 = (const float*)d_in[9];
  const float* wih2 = (const float*)d_in[10];
  const float* whh2 = (const float*)d_in[11];
  const float* bih2 = (const float*)d_in[12];
  const float* bhh2 = (const float*)d_in[13];
  const float* wq   = (const float*)d_in[14];
  const float* bq   = (const float*)d_in[15];
  const float* wv   = (const float*)d_in[16];
  const float* locw = (const float*)d_in[17];
  const float* locproj = (const float*)d_in[18];
  const float* vvec = (const float*)d_in[19];
  const float* wf   = (const float*)d_in[20];
  const float* bfb  = (const float*)d_in[21];
  const float* wsv  = (const float*)d_in[22];
  const float* bsv  = (const float*)d_in[23];
  const float* pnw[5]; const float* pnb[5]; const float* png[5]; const float* pnbt[5];
  for (int i = 0; i < 5; ++i) {
    pnw[i]  = (const float*)d_in[24 + i];
    pnb[i]  = (const float*)d_in[29 + i];
    png[i]  = (const float*)d_in[34 + i];
    pnbt[i] = (const float*)d_in[39 + i];
  }

  char* cur = (char*)d_ws;
  auto take = [&](size_t bytes) -> char* {
    char* p = cur;
    cur += (bytes + 255) & ~(size_t)255;
    return p;
  };
  __bf16* W1IMG = (__bf16*)take((size_t)64 * NW1 * 2);
  __bf16* W2IMG = (__bf16*)take((size_t)64 * NW2 * 2);
  __bf16* MWS   = (__bf16*)take((size_t)51200 * 96 * 2);
  __bf16* PW0P  = (__bf16*)take((size_t)256 * 96 * 2);
  __bf16* PW1   = (__bf16*)take((size_t)256 * 256 * 2);
  __bf16* PNWP  = (__bf16*)take((size_t)4587520 * 2);
  __bf16* X0    = (__bf16*)take((size_t)51200 * 256 * 2);
  __bf16* XPRE  = (__bf16*)take((size_t)800 * 64 * 256 * 2);
  float*  VC    = (float*) take((size_t)64 * 200 * 128 * 4);
  __bf16* H1B0  = (__bf16*)take((size_t)64 * 1024 * 2);
  __bf16* H1B1  = (__bf16*)take((size_t)64 * 1024 * 2);
  __bf16* H2B0  = (__bf16*)take((size_t)64 * 1024 * 2);
  __bf16* H2B1  = (__bf16*)take((size_t)64 * 1024 * 2);
  __bf16* CTXBF = (__bf16*)take((size_t)64 * 512 * 2);
  float*  CTXF  = (float*) take((size_t)64 * 512 * 4);
  int*    BAR   = (int*)   take(256);
  __bf16* PNIN  = (__bf16*)take((size_t)B_ * TP_ * 128 * 2);
  __bf16* PB0   = (__bf16*)take((size_t)B_ * TP_ * 512 * 2);
  __bf16* PB1   = (__bf16*)take((size_t)B_ * TP_ * 512 * 2);

  float* out_mel  = (float*)d_out;
  float* out_res  = out_mel + (size_t)B_ * T_ * M_;
  float* out_stop = out_res + (size_t)B_ * T_ * M_;
  float* out_aw   = out_stop + (size_t)B_ * T_;

  // ---- weight prep: per-block LDS fragment images + bf16 packs ----
  k_img_w1<<<(int)(((size_t)64 * NW1 + 255) / 256), 256, 0, stream>>>(wih1, whh1, W1IMG);
  k_img_w2<<<(int)(((size_t)64 * NW2 + 255) / 256), 256, 0, stream>>>(wih2, whh2, W2IMG);
  k_build_mws<<<(51200 * 96 + 255) / 256, 256, 0, stream>>>(mels, MWS);
  k_pad_w0<<<(256 * 96 + 255) / 256, 256, 0, stream>>>(pre_w0, PW0P);
  k_cvt_bf16<<<(65536 + 255) / 256, 256, 0, stream>>>(pre_w1, PW1, 65536);
  // postnet packed weights: L0 512x640, L1-3 512x2560, L4 128x2560
  const size_t pnoff[5] = {0, 327680, 327680 + 1310720, 327680 + 2 * 1310720,
                           327680 + 3 * 1310720};
  k_pack_pnw<<<(327680 + 255) / 256, 256, 0, stream>>>(pnw[0], PNWP + pnoff[0], 80, 128, 512, 512);
  k_pack_pnw<<<(1310720 + 255) / 256, 256, 0, stream>>>(pnw[1], PNWP + pnoff[1], 512, 512, 512, 512);
  k_pack_pnw<<<(1310720 + 255) / 256, 256, 0, stream>>>(pnw[2], PNWP + pnoff[2], 512, 512, 512, 512);
  k_pack_pnw<<<(1310720 + 255) / 256, 256, 0, stream>>>(pnw[3], PNWP + pnoff[3], 512, 512, 512, 512);
  k_pack_pnw<<<(327680 + 255) / 256, 256, 0, stream>>>(pnw[4], PNWP + pnoff[4], 512, 512, 80, 128);

  // ---- zero recurrent state, barrier, padded postnet buffers ----
  hipMemsetAsync(H1B0, 0, (size_t)64 * 1024 * 2, stream);
  hipMemsetAsync(H1B1, 0, (size_t)64 * 1024 * 2, stream);
  hipMemsetAsync(H2B0, 0, (size_t)64 * 1024 * 2, stream);
  hipMemsetAsync(H2B1, 0, (size_t)64 * 1024 * 2, stream);
  hipMemsetAsync(CTXBF, 0, (size_t)64 * 512 * 2, stream);
  hipMemsetAsync(CTXF, 0, (size_t)64 * 512 * 4, stream);
  hipMemsetAsync(BAR, 0, 256, stream);
  hipMemsetAsync(PNIN, 0, (size_t)B_ * TP_ * 128 * 2, stream);
  hipMemsetAsync(PB0, 0, (size_t)B_ * TP_ * 512 * 2, stream);
  hipMemsetAsync(PB1, 0, (size_t)B_ * TP_ * 512 * 2, stream);

  // ---- prenet + value cache ----
  k_prenet1<<<dim3(800, 4), 256, 0, stream>>>(MWS, PW0P, X0);
  k_prenet2<<<dim3(800, 4), 256, 0, stream>>>(X0, PW1, XPRE);
  k_vcache<<<(B_ * S_ * A_ + 255) / 256, 256, 0, stream>>>(enc, wv, VC);

  // ---- persistent recurrent decode ----
  const size_t smem_bytes = (size_t)NW2 * 2          // LSTM2 weight slice (max)
                          + (size_t)B_ * GS * 4      // gate staging
                          + (size_t)B_ * 16 * 4;     // cell state
  k_decode<<<NBLK, 256, smem_bytes, stream>>>(
      W1IMG, W2IMG, XPRE, bih1, bhh1, bih2, bhh2,
      H1B0, H1B1, H2B0, H2B1, CTXBF, CTXF,
      enc, emask, VC, wq, bq, locw, locproj, vvec,
      wf, bfb, wsv, bsv, out_mel, out_stop, out_aw, BAR, BAR + 1);

  // ---- postnet ----
  k_mel2bf_pad<<<((B_ * T_ * M_) + 255) / 256, 256, 0, stream>>>(out_mel, PNIN);
  k_post<<<dim3(13, 8, 64), 256, 0, stream>>>(PNIN, PNWP + pnoff[0], pnb[0], png[0], pnbt[0],
                                              128, 7, 512, 20, 1, PB0, nullptr);
  k_post<<<dim3(13, 8, 64), 256, 0, stream>>>(PB0, PNWP + pnoff[1], pnb[1], png[1], pnbt[1],
                                              512, 9, 512, 80, 1, PB1, nullptr);
  k_post<<<dim3(13, 8, 64), 256, 0, stream>>>(PB1, PNWP + pnoff[2], pnb[2], png[2], pnbt[2],
                                              512, 9, 512, 80, 1, PB0, nullptr);
  k_post<<<dim3(13, 8, 64), 256, 0, stream>>>(PB0, PNWP + pnoff[3], pnb[3], png[3], pnbt[3],
                                              512, 9, 512, 80, 1, PB1, nullptr);
  k_post<<<dim3(13, 2, 64), 256, 0, stream>>>(PB1, PNWP + pnoff[4], pnb[4], png[4], pnbt[4],
                                              512, 9, 80, 80, 0, nullptr, out_res);

  // ---- masking epilogue ----
  k_mask<<<(B_ * T_ + 255) / 256, 256, 0, stream>>>(dmask, out_res, out_stop, out_aw);
}
